// CIFAR10Net_73014444032591
// MI455X (gfx1250) — matrix operand order repaired and checked by reading, not compiled, not run on hardware
//
#include <hip/hip_runtime.h>

typedef __attribute__((ext_vector_type(16))) _Float16 v16h;
typedef __attribute__((ext_vector_type(8)))  float    v8f;

#define BT 480   // B*T = 32*15

// ---------------- input pad + f32->f16 convert ----------------
// src (480,18,32,32) f32 -> dst (480,18,36,36) f16 (zero pad 2)
__global__ void k_pad_input(const float* __restrict__ src, _Float16* __restrict__ dst) {
  long idx = (long)blockIdx.x * blockDim.x + threadIdx.x;
  const long total = (long)BT * 18 * 36 * 36;
  if (idx >= total) return;
  int x = (int)(idx % 36); long t1 = idx / 36;
  int y = (int)(t1 % 36);  long t2 = t1 / 36;
  int c = (int)(t2 % 18);  int bt = (int)(t2 / 18);
  float v = 0.f;
  if (x >= 2 && x < 34 && y >= 2 && y < 34)
    v = src[(((long)bt * 18 + c) * 32 + (y - 2)) * 32 + (x - 2)];
  dst[idx] = (_Float16)v;
}

// ---------------- weight pack: f32 (M,K) -> f16 (Mpad,Kpad) ----------------
__global__ void k_pack_w(const float* __restrict__ w, _Float16* __restrict__ a,
                         int M, int K, int Mpad, int Kpad) {
  long idx = (long)blockIdx.x * blockDim.x + threadIdx.x;
  long total = (long)Mpad * Kpad;
  if (idx >= total) return;
  int k = (int)(idx % Kpad); int m = (int)(idx / Kpad);
  float v = (m < M && k < K) ? w[(long)m * K + k] : 0.f;
  a[idx] = (_Float16)v;
}

// ---------------- max-pool (binary spikes) + zero pad ----------------
__global__ void k_pool_pad(const _Float16* __restrict__ in, _Float16* __restrict__ out,
                           int C, int Hi, int Wi, int kk, int ss, int pad,
                           int Hp, int Wp, int pooledH, int pooledW) {
  long idx = (long)blockIdx.x * blockDim.x + threadIdx.x;
  long total = (long)BT * C * Hp * Wp;
  if (idx >= total) return;
  int x = (int)(idx % Wp); long t1 = idx / Wp;
  int y = (int)(t1 % Hp);  long t2 = t1 / Hp;
  int c = (int)(t2 % C);   int bt = (int)(t2 / C);
  int px = x - pad, py = y - pad;
  float mv = 0.f;
  if (px >= 0 && px < pooledW && py >= 0 && py < pooledH) {
    const _Float16* img = in + ((long)bt * C + c) * Hi * Wi;
    int y0 = py * ss, x0 = px * ss;
    for (int dy = 0; dy < kk; ++dy)
      for (int dx = 0; dx < kk; ++dx)
        mv = fmaxf(mv, (float)img[(y0 + dy) * Wi + (x0 + dx)]);
  }
  out[idx] = (_Float16)mv;
}

// ---------------- implicit-GEMM conv via v_wmma_f32_16x16x32_f16 ----------------
// One workgroup (8 waves) per (b,t) image. Image + im2col offset table staged
// in LDS. Each wave computes a 16(M)x32(N) register-blocked unit: one A
// fragment + one offset fetch feed two B fragments and two WMMAs per k-step.
// Lanes with n>=Nout or k>=Ktrue need no masking: unstored C columns are
// independent, and padded-A rows are zero so finite garbage B contributes 0.
__global__ void k_conv_wmma(const _Float16* __restrict__ in,
                            const _Float16* __restrict__ Aw,
                            _Float16* __restrict__ spkOut,
                            float* __restrict__ potOut,
                            int C, int Hi, int Wi, int M, int Ho, int Wo,
                            int R, int S, int Ktrue, int Kpad,
                            int Mtiles, int Ntiles, float thr) {
  extern __shared__ char smemRaw[];
  int*      offs = (int*)smemRaw;                           // [Kpad]
  _Float16* simg = (_Float16*)(smemRaw + (size_t)Kpad * 4); // [C*Hi*Wi]

  const int tid  = threadIdx.x;
  const int lane = tid & 31;
  const int wave = tid >> 5;
  const int bt   = blockIdx.x;
  const int Nout = Ho * Wo;
  const int RS   = R * S;
  const int nWaves = blockDim.x >> 5;

  // Build im2col offset table (k -> spatial offset within image)
  for (int k = tid; k < Kpad; k += blockDim.x) {
    int o = 0;
    if (k < Ktrue) {
      int c = k / RS; int rem = k - c * RS;
      int r = rem / S; int s = rem - r * S;
      o = (c * Hi + r) * Wi + s;
    }
    offs[k] = o;
  }
  // Cooperative image load (dword granular; C*Hi*Wi is even for all layers)
  {
    const unsigned elemsU = (unsigned)(C * Hi * Wi) >> 1;
    const unsigned* gsrc = (const unsigned*)(in + (long)bt * C * Hi * Wi);
    unsigned* ldst = (unsigned*)simg;
    for (unsigned i = tid; i < elemsU; i += blockDim.x) ldst[i] = gsrc[i];
  }
  __syncthreads();

  // Per-lane constants for WMMA fragment addressing
  const int aoff = (lane < 16) ? 0 : 8;    // A: K sub-block select
  const int koff = (lane < 16) ? 0 : 16;   // unused; B offsets use aoff
  const int lcol = lane & 15;

  const int nPairs = (Ntiles + 1) >> 1;    // 16x32 units along N
  const int nUnits = Mtiles * nPairs;

  for (int unit = wave; unit < nUnits; unit += nWaves) {
    const int mt = unit / nPairs, pr = unit % nPairs;
    const int m0 = mt * 16, n0 = pr * 32;
    const int nA = n0 + lcol;
    const int nB = n0 + 16 + lcol;
    const bool vA = (nA < Nout), vB = (nB < Nout);
    const int obaseA = vA ? (nA / Wo) * Wi + (nA % Wo) : 0;
    const int obaseB = vB ? (nB / Wo) * Wi + (nB % Wo) : 0;
    const int mA = m0 + lcol;
    const _Float16* arowBase = Aw + (size_t)mA * Kpad + aoff;

    v8f accA = {}, accB = {};
#pragma unroll 2
    for (int ks = 0; ks < Kpad; ks += 32) {
      // A fragment: 8 dwords (two 16B runs -> global_load_b128 pairs)
      union { v16h h; unsigned u[8]; } a;
      const unsigned* arowU = (const unsigned*)(arowBase + ks);
#pragma unroll
      for (int j = 0; j < 8; ++j) a.u[j] = arowU[(j < 4) ? j : j + 4];
      __builtin_prefetch(arowBase + ks + 64, 0, 0);   // next A block -> L2/L0

      // B offsets: two 8-int runs from LDS (k = 8h..8h+7, 16+8h..16+8h+7;
      // 4x ds_load_b128), shared by both N half-tiles
      int ob[16];
      const int4* otab = (const int4*)(offs + ks + aoff);
#pragma unroll
      for (int j = 0; j < 4; ++j) ((int4*)ob)[j] = otab[(j < 2) ? j : j + 2];

      v16h bA, bB;
#pragma unroll
      for (int h = 0; h < 16; ++h) {
        bA[h] = simg[ob[h] + obaseA];
        bB[h] = simg[ob[h] + obaseB];
      }

      accA = __builtin_amdgcn_wmma_f32_16x16x32_f16(false, a.h, false, bA,
                                                    (short)0, accA, false, false);
      accB = __builtin_amdgcn_wmma_f32_16x16x32_f16(false, a.h, false, bB,
                                                    (short)0, accB, false, false);
    }

    // C/D layout: lane&15 = N, VGPR r -> M = r (lanes 0-15) / 8+r (lanes 16-31)
    const int mrow0 = m0 + ((lane < 16) ? 0 : 8);
#pragma unroll
    for (int r = 0; r < 8; ++r) {
      int m = mrow0 + r;
      if (m < M) {
        long rowo = ((long)bt * M + m) * Nout;
        if (vA) {
          if (spkOut) spkOut[rowo + nA] = (accA[r] > thr) ? (_Float16)1.f : (_Float16)0.f;
          else        potOut[rowo + nA] = accA[r];
        }
        if (vB) {
          if (spkOut) spkOut[rowo + nB] = (accB[r] > thr) ? (_Float16)1.f : (_Float16)0.f;
          else        potOut[rowo + nB] = accB[r];
        }
      }
    }
  }
}

// ---------------- winner class: argmax of pot3[:,14] where pot>0 ----------------
__global__ void k_winner(const float* __restrict__ pot3, float* __restrict__ cls) {
  int b = blockIdx.x;
  const float* p = pot3 + (((long)b * 15 + 14) * 200) * 16;
  __shared__ float svals[256];
  __shared__ int   sidx[256];
  float best = 0.f; int bidx = 0x7fffffff;
  for (int i = threadIdx.x; i < 3200; i += blockDim.x) {
    float v = p[i];
    if (v > 0.f && (v > best || (v == best && i < bidx))) { best = v; bidx = i; }
  }
  svals[threadIdx.x] = best; sidx[threadIdx.x] = bidx;
  __syncthreads();
  for (int s = blockDim.x / 2; s > 0; s >>= 1) {
    if ((int)threadIdx.x < s) {
      float ov = svals[threadIdx.x + s]; int oi = sidx[threadIdx.x + s];
      if (ov > svals[threadIdx.x] ||
          (ov == svals[threadIdx.x] && oi < sidx[threadIdx.x])) {
        svals[threadIdx.x] = ov; sidx[threadIdx.x] = oi;
      }
    }
    __syncthreads();
  }
  if (threadIdx.x == 0)
    cls[b] = (svals[0] > 0.f) ? (float)((sidx[0] / 16) / 20) : -1.f;
}

extern "C" void kernel_launch(void* const* d_in, const int* in_sizes, int n_in,
                              void* d_out, int out_size, void* d_ws, size_t ws_size,
                              hipStream_t stream) {
  (void)in_sizes; (void)n_in; (void)out_size; (void)ws_size;
  const float* inp = (const float*)d_in[0];
  const float* w1  = (const float*)d_in[1];
  const float* w2  = (const float*)d_in[2];
  const float* w3  = (const float*)d_in[3];
  float* outF   = (float*)d_out;
  float* clsOut = outF;          // (32,) class as float
  float* pot3   = outF + 32;     // (32,15,200,4,4) f32

  char* base = (char*)d_ws;
  size_t off = 0;
  auto take = [&](size_t bytes) -> char* {
    char* p = base + off;
    off += (bytes + 255) & ~(size_t)255;
    return p;
  };
  _Float16* in1p = (_Float16*)take((size_t)BT * 18 * 36 * 36 * 2); // 22.4 MB
  _Float16* w1h  = (_Float16*)take((size_t)96 * 896 * 2);
  _Float16* w2h  = (_Float16*)take((size_t)256 * 2272 * 2);
  _Float16* w3h  = (_Float16*)take((size_t)208 * 6272 * 2);
  _Float16* spk1 = (_Float16*)take((size_t)BT * 90 * 900 * 2);     // 77.8 MB
  _Float16* p1   = (_Float16*)take((size_t)BT * 90 * 17 * 17 * 2); // 25 MB
  _Float16* spk2 = spk1;  // reuse: spk2 needs 40.6 MB <= 77.8 MB
  _Float16* p2   = p1;    // reuse: p2 needs 15.4 MB <= 25 MB

  { long t = (long)BT * 18 * 36 * 36;
    k_pad_input<<<dim3((unsigned)((t + 255) / 256)), dim3(256), 0, stream>>>(inp, in1p); }
  { long t = (long)96 * 896;
    k_pack_w<<<dim3((unsigned)((t + 255) / 256)), dim3(256), 0, stream>>>(w1, w1h, 90, 882, 96, 896); }
  { long t = (long)256 * 2272;
    k_pack_w<<<dim3((unsigned)((t + 255) / 256)), dim3(256), 0, stream>>>(w2, w2h, 250, 2250, 256, 2272); }
  { long t = (long)208 * 6272;
    k_pack_w<<<dim3((unsigned)((t + 255) / 256)), dim3(256), 0, stream>>>(w3, w3h, 200, 6250, 208, 6272); }

  // conv1: (18,36,36)->(90,30,30), K 882->896, fire @ 15
  { unsigned shmem = 896u * 4 + 18u * 36 * 36 * 2;   // 50240 B
    k_conv_wmma<<<dim3(BT), dim3(256), shmem, stream>>>(
        in1p, w1h, spk1, (float*)nullptr,
        18, 36, 36, 90, 30, 30, 7, 7, 882, 896, 6, 57, 15.0f); }
  // pool 2x2 s2 + pad1: (90,30,30)->(90,17,17)
  { long t = (long)BT * 90 * 17 * 17;
    k_pool_pad<<<dim3((unsigned)((t + 255) / 256)), dim3(256), 0, stream>>>(
        spk1, p1, 90, 30, 30, 2, 2, 1, 17, 17, 15, 15); }
  // conv2: (90,17,17)->(250,13,13), K 2250->2272, fire @ 10
  { unsigned shmem = 2272u * 4 + 90u * 17 * 17 * 2;  // 61108 B
    k_conv_wmma<<<dim3(BT), dim3(256), shmem, stream>>>(
        p1, w2h, spk2, (float*)nullptr,
        90, 17, 17, 250, 13, 13, 5, 5, 2250, 2272, 16, 11, 10.0f); }
  // pool 3x3 s3 + pad2: (250,13,13)->(250,8,8)
  { long t = (long)BT * 250 * 8 * 8;
    k_pool_pad<<<dim3((unsigned)((t + 255) / 256)), dim3(256), 0, stream>>>(
        spk2, p2, 250, 13, 13, 3, 3, 2, 8, 8, 4, 4); }
  // conv3: (250,8,8)->(200,4,4), K 6250->6272, fp32 potentials to d_out
  { unsigned shmem = 6272u * 4 + 250u * 8 * 8 * 2;   // 57088 B
    k_conv_wmma<<<dim3(BT), dim3(256), shmem, stream>>>(
        p2, w3h, (_Float16*)nullptr, pot3,
        250, 8, 8, 200, 4, 4, 5, 5, 6250, 6272, 13, 1, 0.0f); }
  // winner-take-all classification
  k_winner<<<dim3(32), dim3(256), 0, stream>>>(pot3, clsOut);
}
